// TAGConvNet_34342558499547
// MI455X (gfx1250) — compile-verified, matrix operand order checked by reading, not run on hardware
//
#include <hip/hip_runtime.h>

typedef __attribute__((ext_vector_type(16))) _Float16 v16h;
typedef __attribute__((ext_vector_type(8)))  _Float16 v8h;
typedef __attribute__((ext_vector_type(4)))  _Float16 v4h;
typedef __attribute__((ext_vector_type(8)))  float    v8f;

static constexpr int kN = 50000;   // nodes
static constexpr int kE = 800000;  // edges
static constexpr int kF = 128;     // hidden width

// ---------------- workspace layout (bytes, all 256B aligned) ----------------
static constexpr size_t SZ32 = (size_t)kN * kF * 4;   // 25,600,000
static constexpr size_t SZ16 = (size_t)kN * kF * 2;   // 12,800,000
static constexpr size_t OFF_DEG  = 0;                         // kN*4 -> pad
static constexpr size_t OFF_NORM = 200192;                    // kE*4
static constexpr size_t OFF_A    = OFF_NORM + (size_t)kE * 4; // f32 ping
static constexpr size_t OFF_B    = OFF_A + SZ32;              // f32 pong
static constexpr size_t OFF_H0   = OFF_B + SZ32;              // f16 hops
static constexpr size_t OFF_H1   = OFF_H0 + SZ16;
static constexpr size_t OFF_H2   = OFF_H1 + SZ16;
static constexpr size_t OFF_H3   = OFF_H2 + SZ16;
static constexpr size_t OFF_WP   = OFF_H3 + SZ16;             // 10*16384 f16

// ---------------------------- utility kernels -------------------------------
__global__ void zero_f32x4(float4* __restrict__ p, int n4) {
  int i = blockIdx.x * blockDim.x + threadIdx.x;
  if (i < n4) p[i] = make_float4(0.f, 0.f, 0.f, 0.f);
}

__global__ void cvt_f16x4(const float4* __restrict__ in, v4h* __restrict__ out, int n4) {
  int i = blockIdx.x * blockDim.x + threadIdx.x;
  if (i < n4) {
    float4 v = in[i];
    v4h h;
    h[0] = (_Float16)v.x; h[1] = (_Float16)v.y;
    h[2] = (_Float16)v.z; h[3] = (_Float16)v.w;
    out[i] = h;
  }
}

// --------------------------- graph normalization ----------------------------
__global__ void deg_kernel(const int* __restrict__ col, float* __restrict__ deg, int e) {
  int i = blockIdx.x * blockDim.x + threadIdx.x;
  if (i < e) atomicAdd(&deg[col[i]], 1.0f);
}

__global__ void dis_kernel(float* __restrict__ deg, int n) {
  int i = blockIdx.x * blockDim.x + threadIdx.x;
  if (i < n) { float d = deg[i]; deg[i] = d > 0.f ? rsqrtf(d) : 0.f; }
}

__global__ void norm_kernel(const int* __restrict__ row, const int* __restrict__ col,
                            const float* __restrict__ dis, float* __restrict__ nrm, int e) {
  int i = blockIdx.x * blockDim.x + threadIdx.x;
  if (i < e) nrm[i] = dis[row[i]] * dis[col[i]];
}

// ------------------------- sparse propagation (A·h) -------------------------
// out[col[e]][f] += norm[e] * h[row[e]][f] ; one thread per (edge, 4 features)
__global__ void prop_kernel(const float* __restrict__ hin, float* __restrict__ hout,
                            const int* __restrict__ row, const int* __restrict__ col,
                            const float* __restrict__ nrm, int e) {
  long long t = (long long)blockIdx.x * blockDim.x + threadIdx.x;
  int ed = (int)(t >> 5);
  int f4 = (int)(t & 31) << 2;
  if (ed < e) {
    float nv = nrm[ed];
    const float4 v = *(const float4*)(hin + (size_t)row[ed] * kF + f4);
    float* dst = hout + (size_t)col[ed] * kF + f4;
    atomicAdd(dst + 0, nv * v.x);
    atomicAdd(dst + 1, nv * v.y);
    atomicAdd(dst + 2, nv * v.z);
    atomicAdd(dst + 3, nv * v.w);
  }
}

// --------------------- weight pack to WMMA B-fragment -----------------------
// B fragment (16x16x32 f16): lane L -> n = nn*16 + (L&15), g = L>>4;
// halves e<8 : K = kk*32 + g*8 + e ; halves e>=8 : K = kk*32 + 16 + g*8 + (e-8)
// Packed: wp[mat][kk][nn][lane][16 halves], contiguous 32B per lane.
__global__ void pack_w(const float* __restrict__ w, _Float16* __restrict__ wp, int nmats) {
  int t = blockIdx.x * blockDim.x + threadIdx.x;
  if (t >= nmats * 16384) return;
  int mat = t / 16384, r = t % 16384;
  int kk = r >> 12;        r &= 4095;
  int nn = r >> 9;         r &= 511;
  int lane = r >> 4;
  int e = r & 15;
  int g = lane >> 4, ln = lane & 15;
  int k = kk * 32 + (e < 8 ? g * 8 + e : 16 + g * 8 + (e - 8));
  int n = nn * 16 + ln;
  wp[t] = (_Float16)w[(size_t)mat * 16384 + (size_t)k * kF + n];
}

// ------------------- fused multi-matrix WMMA GEMM + bias + relu -------------
// out[M,128] = relu( sum_{mat} H_mat[M,128] @ W_mat[128,128] + bias )
// 8 waves / block, each wave owns 16 rows x 128 cols (8 accumulators of v8f).
// Epilogue writes f32 always and (optionally) an f16 copy for the next layer.
template <int NM>
__global__ __launch_bounds__(256) void gemm_relu(
    const _Float16* __restrict__ h0, const _Float16* __restrict__ h1,
    const _Float16* __restrict__ h2, const _Float16* __restrict__ h3,
    const _Float16* __restrict__ wp, const float* __restrict__ bias,
    float* __restrict__ out32, _Float16* __restrict__ out16, int nrows) {
  const _Float16* hs[4] = {h0, h1, h2, h3};
  int lane = threadIdx.x & 31, wv = threadIdx.x >> 5;
  int m0 = blockIdx.x * 128 + wv * 16;
  if (m0 >= nrows) return;                 // wave-uniform (nrows % 16 == 0)
  int ln = lane & 15, g = lane >> 4;

  v8f acc[8];
#pragma unroll
  for (int nn = 0; nn < 8; nn++) {
    float bv = bias[nn * 16 + ln];
#pragma unroll
    for (int r = 0; r < 8; r++) acc[nn][r] = bv;
  }

#pragma unroll
  for (int mat = 0; mat < NM; mat++) {
    const _Float16* A = hs[mat] + (size_t)(m0 + ln) * kF;
    const _Float16* W = wp + (size_t)mat * 16384 + lane * 16;
#pragma unroll
    for (int kk = 0; kk < 4; kk++) {
      union { v16h v; v8h h[2]; } a;
      // 16-bit A layout: lanes 0-15 K={0..7,16..23}+kk*32 ; lanes 16-31 +8
      a.h[0] = *(const v8h*)(A + kk * 32 + g * 8);
      a.h[1] = *(const v8h*)(A + kk * 32 + 16 + g * 8);
      // preload the 8 independent B fragments so loads clause together and
      // the 8 WMMAs issue without per-op loadcnt==0 stalls
      v16h b[8];
#pragma unroll
      for (int nn = 0; nn < 8; nn++)
        b[nn] = *(const v16h*)(W + kk * 4096 + nn * 512);
#pragma unroll
      for (int nn = 0; nn < 8; nn++)
        acc[nn] = __builtin_amdgcn_wmma_f32_16x16x32_f16(
            false, a.v, false, b[nn], (short)0, acc[nn], false, false);
    }
  }

  // D layout: VGPR r -> M = m0 + g*8 + r ; N = nn*16 + ln
  if (out16 != nullptr) {
#pragma unroll
    for (int nn = 0; nn < 8; nn++) {
#pragma unroll
      for (int r = 0; r < 8; r++) {
        float v = acc[nn][r];
        v = v > 0.f ? v : 0.f;
        size_t idx = (size_t)(m0 + g * 8 + r) * kF + nn * 16 + ln;
        out32[idx] = v;
        out16[idx] = (_Float16)v;
      }
    }
  } else {
#pragma unroll
    for (int nn = 0; nn < 8; nn++) {
#pragma unroll
      for (int r = 0; r < 8; r++) {
        float v = acc[nn][r];
        out32[(size_t)(m0 + g * 8 + r) * kF + nn * 16 + ln] = v > 0.f ? v : 0.f;
      }
    }
  }
}

// ------------------------------ tiny dense layers ---------------------------
__global__ void lin0_kernel(const float* __restrict__ x, const float* __restrict__ w,
                            const float* __restrict__ b, float* __restrict__ out32,
                            _Float16* __restrict__ out16) {
  int n = blockIdx.x, f = threadIdx.x;  // 128 threads
  float acc = b[f];
#pragma unroll
  for (int k = 0; k < 16; k++) acc += x[(size_t)n * 16 + k] * w[k * kF + f];
  acc = fmaxf(acc, 0.f);
  out32[(size_t)n * kF + f] = acc;
  out16[(size_t)n * kF + f] = (_Float16)acc;
}

__global__ void lin3_kernel(const float* __restrict__ h, const float* __restrict__ w,
                            const float* __restrict__ b, float* __restrict__ out, int n) {
  int i = blockIdx.x * blockDim.x + threadIdx.x;
  if (i < n) {
    float acc = b[0];
#pragma unroll 8
    for (int k = 0; k < kF; k++) acc += h[(size_t)i * kF + k] * w[k];
    out[i] = fmaxf(acc, 0.f);
  }
}

// ----------------------------------- host -----------------------------------
extern "C" void kernel_launch(void* const* d_in, const int* in_sizes, int n_in,
                              void* d_out, int out_size, void* d_ws, size_t ws_size,
                              hipStream_t stream) {
  const float* x       = (const float*)d_in[0];
  const int*   ei      = (const int*)d_in[1];
  const int*   row     = ei;
  const int*   col     = ei + kE;
  const float* lin0_w  = (const float*)d_in[2];
  const float* lin0_b  = (const float*)d_in[3];
  const float* conv1_w = (const float*)d_in[4];
  const float* conv1_b = (const float*)d_in[5];
  const float* conv2_w = (const float*)d_in[6];
  const float* conv2_b = (const float*)d_in[7];
  const float* lin1_w  = (const float*)d_in[8];
  const float* lin1_b  = (const float*)d_in[9];
  const float* lin2_w  = (const float*)d_in[10];
  const float* lin2_b  = (const float*)d_in[11];
  const float* lin3_w  = (const float*)d_in[12];
  const float* lin3_b  = (const float*)d_in[13];
  float* out = (float*)d_out;

  char* ws = (char*)d_ws;
  float*    deg  = (float*)(ws + OFF_DEG);
  float*    nrm  = (float*)(ws + OFF_NORM);
  float*    bufA = (float*)(ws + OFF_A);
  float*    bufB = (float*)(ws + OFF_B);
  _Float16* h16[4] = {(_Float16*)(ws + OFF_H0), (_Float16*)(ws + OFF_H1),
                      (_Float16*)(ws + OFF_H2), (_Float16*)(ws + OFF_H3)};
  _Float16* wp = (_Float16*)(ws + OFF_WP);

  const int NF  = kN * kF;                           // 6,400,000
  const int NF4 = NF / 4;
  const unsigned gNF4 = (NF4 + 255) / 256;
  const unsigned gN   = (kN + 255) / 256;
  const unsigned gN4  = (kN / 4 + 255) / 256;
  const unsigned gE   = (kE + 255) / 256;
  const unsigned gPROP = (unsigned)(((long long)kE * 32 + 255) / 256);
  const unsigned gGEMM = (kN + 127) / 128;           // 391; rows%16==0

  // ---- GCN normalization (no self loops): deg over col, dis=rsqrt, norm ----
  zero_f32x4<<<gN4, 256, 0, stream>>>((float4*)deg, kN / 4);
  deg_kernel<<<gE, 256, 0, stream>>>(col, deg, kE);
  dis_kernel<<<gN, 256, 0, stream>>>(deg, kN);
  norm_kernel<<<gE, 256, 0, stream>>>(row, col, deg, nrm, kE);

  // ---- pack all 128x128 weights into WMMA fragment layout (f16) ----
  pack_w<<<(4 * 16384 + 255) / 256, 256, 0, stream>>>(conv1_w, wp + 0 * 16384, 4);
  pack_w<<<(4 * 16384 + 255) / 256, 256, 0, stream>>>(conv2_w, wp + 4 * 16384, 4);
  pack_w<<<(16384 + 255) / 256, 256, 0, stream>>>(lin1_w, wp + 8 * 16384, 1);
  pack_w<<<(16384 + 255) / 256, 256, 0, stream>>>(lin2_w, wp + 9 * 16384, 1);

  // ---- h = relu(x @ lin0_w + b), emit f32 + f16 ----
  lin0_kernel<<<kN, kF, 0, stream>>>(x, lin0_w, lin0_b, bufA, h16[0]);

  // ---- two TAGConv layers: hops (fp32 scatter) + fused 4-mat WMMA GEMM ----
  for (int c = 0; c < 2; c++) {
    const float* cb = (c == 0) ? conv1_b : conv2_b;
    _Float16* cw = wp + (size_t)c * 4 * 16384;
    // hop 1: bufA -> bufB
    zero_f32x4<<<gNF4, 256, 0, stream>>>((float4*)bufB, NF4);
    prop_kernel<<<gPROP, 256, 0, stream>>>(bufA, bufB, row, col, nrm, kE);
    cvt_f16x4<<<gNF4, 256, 0, stream>>>((const float4*)bufB, (v4h*)h16[1], NF4);
    // hop 2: bufB -> bufA
    zero_f32x4<<<gNF4, 256, 0, stream>>>((float4*)bufA, NF4);
    prop_kernel<<<gPROP, 256, 0, stream>>>(bufB, bufA, row, col, nrm, kE);
    cvt_f16x4<<<gNF4, 256, 0, stream>>>((const float4*)bufA, (v4h*)h16[2], NF4);
    // hop 3: bufA -> bufB
    zero_f32x4<<<gNF4, 256, 0, stream>>>((float4*)bufB, NF4);
    prop_kernel<<<gPROP, 256, 0, stream>>>(bufA, bufB, row, col, nrm, kE);
    cvt_f16x4<<<gNF4, 256, 0, stream>>>((const float4*)bufB, (v4h*)h16[3], NF4);
    // out = relu(sum_k h_k @ W_k + b) -> bufA (f32) and h16[0] (f16, next layer)
    gemm_relu<4><<<gGEMM, 256, 0, stream>>>(h16[0], h16[1], h16[2], h16[3],
                                            cw, cb, bufA, h16[0], kN);
  }

  // ---- lin1, lin2 (WMMA), lin3 (scalar dot) ----
  gemm_relu<1><<<gGEMM, 256, 0, stream>>>(h16[0], h16[0], h16[0], h16[0],
                                          wp + 8 * 16384, lin1_b, bufB, h16[1], kN);
  gemm_relu<1><<<gGEMM, 256, 0, stream>>>(h16[1], h16[1], h16[1], h16[1],
                                          wp + 9 * 16384, lin2_b, bufA, nullptr, kN);
  lin3_kernel<<<gN, 256, 0, stream>>>(bufA, lin3_w, lin3_b, out, kN);

  (void)in_sizes; (void)n_in; (void)out_size; (void)ws_size;
}
